// SparseNet_Grand_83399674954374
// MI455X (gfx1250) — matrix operand order repaired and checked by reading, not compiled, not run on hardware
//
#include <hip/hip_runtime.h>
#include <hip/hip_bf16.h>

// ---------------------------------------------------------------------------
// MinCut-pool GNN forward for MI455X (gfx1250, wave32, WMMA).
// All GEMMs use v_wmma_f32_16x16x32_f16 with 1x4 tile register blocking:
// each wave owns a 16x64 output strip, loads one A fragment per k-step
// (2x global_load_b128) and issues 4 back-to-back WMMAs against 4 B panels.
// Edge scatters are f32 atomics (irregular, L2-resident working set).
// ---------------------------------------------------------------------------

typedef __attribute__((ext_vector_type(16))) _Float16 v16h;
typedef __attribute__((ext_vector_type(8)))  _Float16 v8h;
typedef __attribute__((ext_vector_type(8)))  float    v8f;

#define IN_C 64
#define H_C  128
#define C1_C 64
#define C2_C 16
#define PRUNE_THRESH (1.0f/63.0f)

// ---------------------------------------------------------------- utilities
__global__ void k_zero(float* __restrict__ p, int n) {
  int t = blockIdx.x * blockDim.x + threadIdx.x;
  if (t < n) p[t] = 0.f;
}

// x_drop = mask*x; stash f16 copy into Acat[:,64:128]; zero agg/A_s/deg.
__global__ void k_init(const float* __restrict__ x, const float* __restrict__ mask,
                       float* __restrict__ xd, _Float16* __restrict__ Acat,
                       float* __restrict__ agg, float* __restrict__ A_s,
                       float* __restrict__ deg, int N) {
  int t = blockIdx.x * blockDim.x + threadIdx.x;
  if (t >= N * 64) return;
  int n = t >> 6, f = t & 63;
  float v = mask[n] * x[t];
  xd[t] = v;
  Acat[(size_t)n * 128 + 64 + f] = (_Float16)v;
  agg[t] = 0.f;
  A_s[t] = 0.f;
  if (f == 0) deg[n] = 0.f;
}

// Repack weights to f16 in WMMA-fragment ("swizzled") order so the GEMM's
// B-fragment is 16 contiguous halves per lane:
//   Bs[((tn*ntk + kk)*32 + lane)*16 + i] = B[kk*32 + ((i>>3)<<4) + ((lane>>4)<<3) + (i&7)][tn*16 + (lane&15)]
__global__ void k_prepw(const float* __restrict__ W1_rel, const float* __restrict__ W1_root,
                        const float* __restrict__ pW1,
                        _Float16* __restrict__ W1cat_s, _Float16* __restrict__ pW1T_s) {
  int t = blockIdx.x * blockDim.x + threadIdx.x;
  if (t < 128 * 128) {
    int i = t & 15, lane = (t >> 4) & 31, r = t >> 9;
    int kk = r & 3, tn = r >> 2;                       // ntk = 4
    int k = kk * 32 + ((i >> 3) << 4) + ((lane >> 4) << 3) + (i & 7);
    int n = tn * 16 + (lane & 15);
    float v = (k < 64) ? W1_rel[n * 64 + k] : W1_root[n * 64 + (k - 64)];
    W1cat_s[t] = (_Float16)v;
  } else if (t < 128 * 128 + 128 * 64) {
    int t2 = t - 128 * 128;
    int i = t2 & 15, lane = (t2 >> 4) & 31, r = t2 >> 9;
    int kk = r & 3, tn = r >> 2;                       // ntk = 4, 4 col tiles
    int k = kk * 32 + ((i >> 3) << 4) + ((lane >> 4) << 3) + (i & 7);
    int c = tn * 16 + (lane & 15);
    pW1T_s[t2] = (_Float16)pW1[c * 128 + k];
  }
}

// dst[si[e]] += src[gi[e]] over 64 feats; one (edge,lane) pair handles 2 feats.
__global__ void k_scatter(const int* __restrict__ gi, const int* __restrict__ si,
                          const float* __restrict__ src, float* __restrict__ dst,
                          float* __restrict__ deg, int E) {
  long total = (long)E * 32;
  for (long t = (long)blockIdx.x * blockDim.x + threadIdx.x; t < total;
       t += (long)gridDim.x * blockDim.x) {
    int e = (int)(t >> 5);
    int lane = (int)(t & 31);
    int g = gi[e], s = si[e];
    int f = lane << 1;
    float2 v = *(const float2*)(src + (size_t)g * 64 + f);
    atomicAdd(dst + (size_t)s * 64 + f, v.x);
    atomicAdd(dst + (size_t)s * 64 + f + 1, v.y);
    if (deg && lane == 0) atomicAdd(deg + g, 1.0f);
  }
}

// agg f32 -> f16 into Acat[:,0:64]
__global__ void k_cvt_agg(const float* __restrict__ agg, _Float16* __restrict__ Acat, int N) {
  int t = blockIdx.x * blockDim.x + threadIdx.x;
  if (t >= N * 64) return;
  int n = t >> 6, f = t & 63;
  Acat[(size_t)n * 128 + f] = (_Float16)agg[t];
}

// f32 [N, C] row-major -> f16 transposed [C, Npad] with zero padding.
__global__ void k_cvtT(const float* __restrict__ src, _Float16* __restrict__ dstT,
                       int N, int Npad, int C) {
  int t = blockIdx.x * blockDim.x + threadIdx.x;
  if (t >= C * Npad) return;
  int c = t / Npad, n = t - c * Npad;
  dstT[t] = (n < N) ? (_Float16)src[(size_t)n * C + c] : (_Float16)0.f;
}

// zero the pad columns of a transposed f16 buffer [C, Npad]
__global__ void k_padzero(_Float16* __restrict__ dstT, int N, int Npad, int C) {
  int pad = Npad - N;
  int t = blockIdx.x * blockDim.x + threadIdx.x;
  if (t >= C * pad) return;
  int c = t / pad, i = t - c * pad;
  dstT[(size_t)c * Npad + N + i] = (_Float16)0.f;
}

// ------------------------------------------------------------- WMMA GEMM
// out = (A[M,K] @ B[K,Ncols]) + bias, optional ReLU; Ncols multiple of 64.
// A row-major f16; B pre-swizzled panels. Each wave: 16x64 strip (4 tiles).
// Optional outputs: C (f32 row-major), Ch (f16 row-major), ChT (f16 [Ncols,ldT]).
__global__ void k_wmma_gemm(const _Float16* __restrict__ A, int lda,
                            const _Float16* __restrict__ Bs,
                            const float* __restrict__ bias,
                            float* __restrict__ C, _Float16* __restrict__ Ch,
                            _Float16* __restrict__ ChT, int ldT,
                            int Mrows, int Ktot, int Ncols, int relu) {
  int lane = threadIdx.x & 31;
  int wave = threadIdx.x >> 5;
  int ntk = Ktot >> 5;
  int ng = Ncols >> 6;                       // groups of 4 column tiles
  int tile = blockIdx.x * (blockDim.x >> 5) + wave;
  int total = (Mrows >> 4) * ng;
  if (tile >= total) return;                 // wave-uniform: EXEC stays all-1
  int tg = tile % ng, tm = tile / ng;
  int row0 = tm << 4, col0 = tg << 6;
  int m = row0 + (lane & 15);
  int ka = (lane >> 4) << 3;                 // A: +0 / +8
  const v8h* pa = (const v8h*)(A + (size_t)m * lda + ka);
  const v8h* pb = (const v8h*)Bs + ((size_t)(tg * 4) * ntk * 32 + lane) * 2;
  size_t pstride = (size_t)ntk * 64;         // v8h stride between column panels
  v8f acc[4] = {};
  for (int kk = 0; kk < ntk; ++kk) {
    v8h a_lo = pa[0];
    v8h a_hi = pa[2];                        // +16 halves
    pa += 4;
    v16h af;
#pragma unroll
    for (int i = 0; i < 8; ++i) { af[i] = a_lo[i]; af[8 + i] = a_hi[i]; }
#pragma unroll
    for (int j = 0; j < 4; ++j) {
      const v8h* pbj = pb + (size_t)j * pstride;
      v8h b_lo = pbj[0];
      v8h b_hi = pbj[1];
      v16h bf;
#pragma unroll
      for (int i = 0; i < 8; ++i) { bf[i] = b_lo[i]; bf[8 + i] = b_hi[i]; }
      acc[j] = __builtin_amdgcn_wmma_f32_16x16x32_f16(false, af, false, bf,
                                                      (short)0, acc[j], false, false);
    }
    pb += 64;
  }
  int mb = row0 + ((lane >> 4) << 3);
#pragma unroll
  for (int j = 0; j < 4; ++j) {
    int n = col0 + j * 16 + (lane & 15);
    float bn = bias ? bias[n] : 0.f;
    v8h hv;
#pragma unroll
    for (int r = 0; r < 8; ++r) {
      float v = acc[j][r] + bn;
      if (relu) v = fmaxf(v, 0.f);
      size_t idx = (size_t)(mb + r) * Ncols + n;
      if (C)  C[idx]  = v;
      if (Ch) Ch[idx] = (_Float16)v;
      hv[r] = (_Float16)v;
    }
    if (ChT) *(v8h*)(ChT + (size_t)n * ldT + mb) = hv;   // 8 contiguous m's
  }
}

// Split-K reduction GEMM on transposed f16 operands:
//   C[64,Ncols] += At[64,Kpad] x Bt[Ncols,Kpad]^T  (C[m][n] += sum_k At[m][k]*Bt[n][k])
// Kpad multiple of 32, pads zero. Each wave: 16x64 strip (4 tiles), split-K atomics.
__global__ void k_wmma_red(const _Float16* __restrict__ At, int lda,
                           const _Float16* __restrict__ Bt, int ldb,
                           float* __restrict__ C, int Ncols,
                           int Kpad, int kchunk) {
  int lane = threadIdx.x & 31;
  int wave = threadIdx.x >> 5;
  int ng = Ncols >> 6;
  int tile = blockIdx.x * (blockDim.x >> 5) + wave;
  int total = 4 * ng;                        // M = 64 clusters
  if (tile >= total) return;
  int tg = tile % ng, tm = tile / ng;
  int row0 = tm << 4, col0 = tg << 6;
  int m = row0 + (lane & 15);
  int ka = (lane >> 4) << 4;                 // A: +0 / +16
  int kb = (lane >> 4) << 4;                 // B: +0 / +16
  int kbeg = blockIdx.y * kchunk;
  int kend = kbeg + kchunk; if (kend > Kpad) kend = Kpad;
  const v8h* pa = (const v8h*)(At + (size_t)m * lda + kbeg + ka);
  const v8h* pb[4];
#pragma unroll
  for (int j = 0; j < 4; ++j)
    pb[j] = (const v8h*)(Bt + (size_t)(col0 + j * 16 + (lane & 15)) * ldb + kbeg + kb);
  v8f acc[4] = {};
  for (int k0 = kbeg; k0 < kend; k0 += 32) {
    v8h a_lo = pa[0];
    v8h a_hi = pa[1];
    pa += 4;
    v16h af;
#pragma unroll
    for (int i = 0; i < 8; ++i) { af[i] = a_lo[i]; af[8 + i] = a_hi[i]; }
#pragma unroll
    for (int j = 0; j < 4; ++j) {
      v8h b_lo = pb[j][0];
      v8h b_hi = pb[j][1];
      pb[j] += 4;
      v16h bf;
#pragma unroll
      for (int i = 0; i < 8; ++i) { bf[i] = b_lo[i]; bf[8 + i] = b_hi[i]; }
      acc[j] = __builtin_amdgcn_wmma_f32_16x16x32_f16(false, af, false, bf,
                                                      (short)0, acc[j], false, false);
    }
  }
  int mb = row0 + ((lane >> 4) << 3);
#pragma unroll
  for (int j = 0; j < 4; ++j) {
    int n = col0 + j * 16 + (lane & 15);
#pragma unroll
    for (int r = 0; r < 8; ++r)
      atomicAdd(&C[(size_t)(mb + r) * Ncols + n], acc[j][r]);
  }
}

// --------------------------------------------- per-row LN + softmax (wave32)
__global__ void k_rownorm(const float* __restrict__ s1lin, const float* __restrict__ g,
                          const float* __restrict__ be, const float* __restrict__ deg,
                          float* __restrict__ ls_out, float* __restrict__ s_out,
                          float* __restrict__ denp, int N) {
  int wid = (int)(((long)blockIdx.x * blockDim.x + threadIdx.x) >> 5);
  int lane = threadIdx.x & 31;
  if (wid >= N) return;
  size_t base = (size_t)wid * 64;
  float x0 = s1lin[base + lane * 2], x1 = s1lin[base + lane * 2 + 1];
  float sm = x0 + x1, sq = x0 * x0 + x1 * x1;
  for (int o = 16; o; o >>= 1) { sm += __shfl_xor(sm, o, 32); sq += __shfl_xor(sq, o, 32); }
  float mean = sm * 0.015625f;
  float var = sq * 0.015625f - mean * mean;
  float inv = rsqrtf(var + 1e-5f);
  int c0 = lane * 2, c1 = c0 + 1;
  float y0 = (x0 - mean) * inv * g[c0] + be[c0];
  float y1 = (x1 - mean) * inv * g[c1] + be[c1];
  float mx = fmaxf(y0, y1);
  for (int o = 16; o; o >>= 1) mx = fmaxf(mx, __shfl_xor(mx, o, 32));
  float es = __expf(y0 - mx) + __expf(y1 - mx);
  for (int o = 16; o; o >>= 1) es += __shfl_xor(es, o, 32);
  float lse = mx + __logf(es);
  float l0 = y0 - lse, l1 = y1 - lse;
  float p0 = __expf(l0), p1 = __expf(l1);
  ls_out[base + c0] = l0; ls_out[base + c1] = l1;
  s_out[base + c0] = p0;  s_out[base + c1] = p1;
  float ps = p0 * p0 + p1 * p1;
  for (int o = 16; o; o >>= 1) ps += __shfl_xor(ps, o, 32);
  if (lane == 0) atomicAdd(denp, deg[wid] * ps);
}

// ------------------------------------------------------------- finale block
__device__ inline float blockReduce(float v, float* red) {
  int t = threadIdx.x;
  red[t] = v;
  __syncthreads();
  for (int s = 128; s > 0; s >>= 1) {
    if (t < s) red[t] += red[t + s];
    __syncthreads();
  }
  float r = red[0];
  __syncthreads();
  return r;
}

__global__ __launch_bounds__(256) void k_final(
    const float* __restrict__ Px, const float* __restrict__ P,
    const float* __restrict__ sTs, const float* __restrict__ denp,
    const float* __restrict__ W2_rel, const float* __restrict__ b2,
    const float* __restrict__ W2_root, const float* __restrict__ pW2,
    const float* __restrict__ pb2, const float* __restrict__ g2,
    const float* __restrict__ be2,
    float* __restrict__ tmpA, float* __restrict__ x2g,
    float* __restrict__ out, float* __restrict__ ls2) {
  __shared__ float Mm[64 * 64];
  __shared__ float dsi[64];
  __shared__ float deg2[64];
  __shared__ float s2[64 * 16];
  __shared__ float As2[64 * 16];
  __shared__ float red[256];
  __shared__ float st2[256];
  int t = threadIdx.x;

  // mc1 / o1 -------------------------------------------------------------
  float a = 0.f;
  for (int i = t; i < 4096; i += 256) a += sTs[i] * sTs[i];
  float norm = sqrtf(blockReduce(a, red));
  float ob = 0.f;
  for (int i = t; i < 4096; i += 256) {
    int r = i >> 6, c = i & 63;
    float v = sTs[i] / (norm + 1e-10f) - ((r == c) ? 0.125f : 0.f);  // I/sqrt(64)
    ob += v * v;
  }
  float osq = blockReduce(ob, red);
  float trc = (t < 64) ? P[t * 65] : 0.f;
  float tr = blockReduce(trc, red);
  if (t == 0) { out[0] = -tr / (denp[0] + 1e-10f); out[1] = sqrtf(osq); }

  // adj_norm -> pruning mask M -------------------------------------------
  if (t < 64) {
    float rs = 0.f;
    for (int j = 0; j < 64; ++j) rs += (j == t) ? 0.f : P[t * 64 + j];
    dsi[t] = 1.f / (sqrtf(rs) + 1e-15f);
  }
  __syncthreads();
  for (int i = t; i < 4096; i += 256) {
    int r = i >> 6, c = i & 63;
    float av = (r == c) ? 0.f : P[i];
    Mm[i] = (av * dsi[r] * dsi[c] > PRUNE_THRESH) ? 1.f : 0.f;
  }
  __syncthreads();

  // tmpA = M^T @ Px  [64,128] --------------------------------------------
  for (int idx = t; idx < 8192; idx += 256) {
    int i = idx >> 7, k = idx & 127;
    float v = 0.f;
    for (int c = 0; c < 64; ++c) v += Mm[c * 64 + i] * Px[c * 128 + k];
    tmpA[idx] = v;
  }
  __threadfence_block();
  __syncthreads();

  // x2 = relu(tmpA @ W2_rel^T + b2 + Px @ W2_root^T) ---------------------
  for (int idx = t; idx < 8192; idx += 256) {
    int i = idx >> 7, j = idx & 127;
    float v = b2[j];
    const float* ta = tmpA + i * 128;
    const float* px = Px + i * 128;
    const float* wr = W2_rel + j * 128;
    const float* wo = W2_root + j * 128;
    for (int k = 0; k < 128; ++k) v += ta[k] * wr[k] + px[k] * wo[k];
    x2g[idx] = fmaxf(v, 0.f);
  }
  __threadfence_block();
  __syncthreads();

  // s2 = softmax(LN(x2 @ pW2^T + pb2)); ls2 = log_softmax; deg2 ----------
  if (t < 64) {
    float lin[16];
    const float* xr = x2g + t * 128;
    float mean = 0.f;
    for (int c = 0; c < 16; ++c) {
      float v = pb2[c];
      const float* pw = pW2 + c * 128;
      for (int k = 0; k < 128; ++k) v += xr[k] * pw[k];
      lin[c] = v; mean += v;
    }
    mean *= 0.0625f;
    float var = 0.f;
    for (int c = 0; c < 16; ++c) { float d = lin[c] - mean; var += d * d; }
    var *= 0.0625f;
    float inv = rsqrtf(var + 1e-5f);
    float y[16], mx = -1e30f;
    for (int c = 0; c < 16; ++c) {
      y[c] = (lin[c] - mean) * inv * g2[c] + be2[c];
      mx = fmaxf(mx, y[c]);
    }
    float es = 0.f;
    for (int c = 0; c < 16; ++c) es += __expf(y[c] - mx);
    float lse = mx + __logf(es);
    for (int c = 0; c < 16; ++c) {
      float l = y[c] - lse;
      ls2[t * 16 + c] = l;
      s2[t * 16 + c] = __expf(l);
    }
    float dd = 0.f;
    for (int j = 0; j < 64; ++j) dd += Mm[t * 64 + j];
    deg2[t] = dd;
  }
  __syncthreads();

  // As2 = M @ s2 ----------------------------------------------------------
  if (t < 64) {
    for (int c = 0; c < 16; ++c) {
      float v = 0.f;
      for (int j = 0; j < 64; ++j) v += Mm[t * 64 + j] * s2[j * 16 + c];
      As2[t * 16 + c] = v;
    }
  }
  __syncthreads();

  // mc2 / o2 --------------------------------------------------------------
  int cc = t >> 4, dd = t & 15;
  float pa = 0.f, st = 0.f;
  for (int i = 0; i < 64; ++i) {
    pa += s2[i * 16 + cc] * As2[i * 16 + dd];
    st += s2[i * 16 + cc] * s2[i * 16 + dd];
  }
  st2[t] = st;
  __syncthreads();
  float num2 = blockReduce((cc == dd) ? pa : 0.f, red);
  float dv = 0.f;
  if (t < 64) {
    float sqs = 0.f;
    for (int c = 0; c < 16; ++c) { float v = s2[t * 16 + c]; sqs += v * v; }
    dv = deg2[t] * sqs;
  }
  float den2 = blockReduce(dv, red);
  float norm2 = sqrtf(blockReduce(st * st, red));
  float w = st2[t] / (norm2 + 1e-10f) - ((cc == dd) ? 0.25f : 0.f);  // I/sqrt(16)
  float o2s = blockReduce(w * w, red);
  if (t == 0) { out[2] = -num2 / (den2 + 1e-10f); out[3] = sqrtf(o2s); }
}

// ---------------------------------------------------------------------------
extern "C" void kernel_launch(void* const* d_in, const int* in_sizes, int n_in,
                              void* d_out, int out_size, void* d_ws, size_t ws_size,
                              hipStream_t stream) {
  const float* x       = (const float*)d_in[0];
  const int*   ei      = (const int*)d_in[1];
  const float* mask    = (const float*)d_in[2];
  const float* W1_rel  = (const float*)d_in[3];
  const float* b1      = (const float*)d_in[4];
  const float* W1_root = (const float*)d_in[5];
  const float* pW1     = (const float*)d_in[6];
  const float* pb1     = (const float*)d_in[7];
  const float* g1      = (const float*)d_in[8];
  const float* be1     = (const float*)d_in[9];
  const float* W2_rel  = (const float*)d_in[10];
  const float* b2      = (const float*)d_in[11];
  const float* W2_root = (const float*)d_in[12];
  const float* pW2     = (const float*)d_in[13];
  const float* pb2     = (const float*)d_in[14];
  const float* g2      = (const float*)d_in[15];
  const float* be2     = (const float*)d_in[16];

  int N = in_sizes[0] / IN_C;           // 50000
  int E = in_sizes[1] / 2;              // 800000
  int Npad = (N + 31) & ~31;            // 50016 (multiple of 32)
  const int* row = ei;
  const int* col = ei + E;

  // workspace carve-out (256B aligned)
  char* w = (char*)d_ws;
  size_t off = 0;
  auto get = [&](size_t bytes) -> char* {
    char* p = w + off;
    off = (off + bytes + 255) & ~(size_t)255;
    return p;
  };
  float*     xd      = (float*)get((size_t)N * 64 * 4);
  float*     agg     = (float*)get((size_t)N * 64 * 4);
  _Float16*  Acat    = (_Float16*)get((size_t)N * 128 * 2);
  _Float16*  x1h     = (_Float16*)get((size_t)N * 128 * 2);
  _Float16*  x1T     = (_Float16*)get((size_t)Npad * 128 * 2);
  float*     s1lin   = (float*)get((size_t)N * 64 * 4);
  float*     sft     = (float*)get((size_t)N * 64 * 4);
  _Float16*  sT      = (_Float16*)get((size_t)Npad * 64 * 2);
  float*     A_s     = (float*)get((size_t)N * 64 * 4);
  _Float16*  A_sT    = (_Float16*)get((size_t)Npad * 64 * 2);
  float*     deg     = (float*)get((size_t)N * 4);
  _Float16*  W1cat_s = (_Float16*)get(128 * 128 * 2);
  _Float16*  pW1T_s  = (_Float16*)get(128 * 64 * 2);
  float*     small   = (float*)get(16400 * 4);   // Px|P|sTs|den contiguous
  float*     Px      = small;
  float*     P       = small + 8192;
  float*     sTsb    = small + 12288;
  float*     denp    = small + 16384;
  float*     tmpA    = (float*)get(8192 * 4);
  float*     x2g     = (float*)get(8192 * 4);

  float* out = (float*)d_out;
  float* ls1 = out + 4;
  float* ls2 = out + 4 + (size_t)N * 64;

  int feat_blocks = (N * 64 + 255) / 256;
  long sc_total = (long)E * 32;
  int sc_blocks = (int)((sc_total + 255) / 256);
  int cvtT_blocks = (64 * Npad + 255) / 256;

  // 0: zero the small accumulators (Px, P, sTs, den)
  k_zero<<<(16385 + 255) / 256, 256, 0, stream>>>(small, 16385);
  // 1: drop-mask, zero agg/A_s/deg, fill Acat[:,64:128]
  k_init<<<feat_blocks, 256, 0, stream>>>(x, mask, xd, Acat, agg, A_s, deg, N);
  // 2: weight f16 repack into fragment-swizzled panels
  k_prepw<<<(128 * 128 + 128 * 64 + 255) / 256, 256, 0, stream>>>(W1_rel, W1_root, pW1,
                                                                  W1cat_s, pW1T_s);
  // 3: agg[col] += xd[row]; deg[row] += 1
  k_scatter<<<sc_blocks, 256, 0, stream>>>(row, col, xd, agg, deg, E);
  // 4: Acat[:,0:64] = f16(agg)
  k_cvt_agg<<<feat_blocks, 256, 0, stream>>>(agg, Acat, N);
  // 5: x1 = relu(Acat @ W1cat + b1) -> x1h row-major + x1T transposed (WMMA)
  {
    int strips = (N / 16) * (128 / 64);        // 6250 wave-strips
    k_wmma_gemm<<<(strips + 3) / 4, 128, 0, stream>>>(Acat, 128, W1cat_s, b1,
                                                      (float*)nullptr, x1h, x1T, Npad,
                                                      N, 128, 128, 1);
  }
  k_padzero<<<(128 * (Npad - N) + 255) / 256, 256, 0, stream>>>(x1T, N, Npad, 128);
  // 6: s1lin = x1 @ pW1^T + pb1 (WMMA)
  {
    int strips = (N / 16) * (64 / 64);         // 3125 wave-strips
    k_wmma_gemm<<<(strips + 3) / 4, 128, 0, stream>>>(x1h, 128, pW1T_s, pb1,
                                                      s1lin, (_Float16*)nullptr,
                                                      (_Float16*)nullptr, 0,
                                                      N, 128, 64, 0);
  }
  // 7: LN + softmax + log_softmax per row; den accumulation
  k_rownorm<<<(N + 3) / 4, 128, 0, stream>>>(s1lin, g1, be1, deg, ls1, sft, denp, N);
  // 8: sT = transpose(f16(sft)) with zero pad
  k_cvtT<<<cvtT_blocks, 256, 0, stream>>>(sft, sT, N, Npad, 64);
  // 9: A_s[row] += s[col]
  k_scatter<<<sc_blocks, 256, 0, stream>>>(col, row, sft, A_s, (float*)nullptr, E);
  // 10: A_sT = transpose(f16(A_s)) with zero pad
  k_cvtT<<<cvtT_blocks, 256, 0, stream>>>(A_s, A_sT, N, Npad, 64);
  // 11-13: split-K WMMA reductions: Px += s^T x1; P += s^T A_s; sTs += s^T s
  {
    int kchunk = 1280;                       // multiple of 32
    int nch = (Npad + kchunk - 1) / kchunk;  // 40
    dim3 gPx(2, nch);                        // 8 strip-tiles / 4 waves per block
    dim3 g64(1, nch);                        // 4 strip-tiles / 4 waves per block
    k_wmma_red<<<gPx, 128, 0, stream>>>(sT, Npad, x1T, Npad, Px, 128, Npad, kchunk);
    k_wmma_red<<<g64, 128, 0, stream>>>(sT, Npad, A_sT, Npad, P, 64, Npad, kchunk);
    k_wmma_red<<<g64, 128, 0, stream>>>(sT, Npad, sT, Npad, sTsb, 64, Npad, kchunk);
  }
  // 14: losses, pruning, pooled GraphConv, second pooling head
  k_final<<<1, 256, 0, stream>>>(Px, P, sTsb, denp, W2_rel, b2, W2_root,
                                 pW2, pb2, g2, be2, tmpA, x2g, out, ls2);
}
